// WaveletDualAttentionBlock_Window_46420006535714
// MI455X (gfx1250) — matrix operand order repaired and checked by reading, not compiled, not run on hardware
//
#include <hip/hip_runtime.h>
#include <hip/hip_bf16.h>
#include <math.h>

typedef __attribute__((ext_vector_type(16))) _Float16 v16h;
typedef __attribute__((ext_vector_type(8)))  _Float16 v8h;
typedef __attribute__((ext_vector_type(8)))  float    v8f;

#define PIX 4096   // 64*64 pixels per batch image
#define NBATCH 8

// ---------------------------------------------------------------------------
// helpers
// ---------------------------------------------------------------------------
__device__ __forceinline__ float gelu_exact(float x) {
    return 0.5f * x * (1.0f + erff(x * 0.70710678118654752440f));
}

// A fragment (16x32 f16, MxK) from row-major LDS tile, leading dim ld (halves,
// multiple of 8 so rows are 16B aligned).
// Layout (ISA 7.12.2): lane L: M=L%16 ; halves 0..7 = K kh..kh+7, 8..15 = K 16+kh..16+kh+7
// with kh = (L>=16)?8:0  -> two contiguous 16B runs per lane.
__device__ __forceinline__ v16h ld_a_frag(const _Float16* base, int ld) {
    int lane = threadIdx.x & 31;
    int m    = lane & 15;
    int kh   = (lane >> 4) << 3;   // 0 or 8
    const _Float16* p = base + m * ld + kh;
    v8h lo = *(const v8h*)(p);
    v8h hi = *(const v8h*)(p + 16);
    return __builtin_shufflevector(lo, hi, 0, 1, 2, 3, 4, 5, 6, 7,
                                   8, 9, 10, 11, 12, 13, 14, 15);
}

// B fragment (32x16 f16, KxN) from a TRANSPOSED LDS tile stored [N][K]
// (leading dim ld halves, multiple of 8). Lane L: N=L%16, kb=(L>=16)?8:0,
// halves 0..7 = K kb..kb+7, 8..15 = K 16+kb..16+kb+7 -> two 16B runs per lane.
__device__ __forceinline__ v16h ld_b_frag_t(const _Float16* baseT, int ld) {
    int lane = threadIdx.x & 31;
    int n    = lane & 15;
    int kb   = (lane >> 4) << 3;   // 0 or 8
    const _Float16* p = baseT + n * ld + kb;
    v8h lo = *(const v8h*)(p);
    v8h hi = *(const v8h*)(p + 16);
    return __builtin_shufflevector(lo, hi, 0, 1, 2, 3, 4, 5, 6, 7,
                                   8, 9, 10, 11, 12, 13, 14, 15);
}

// ---------------------------------------------------------------------------
// small utility kernels
// ---------------------------------------------------------------------------
__global__ void cvt_f16_kernel(const float* __restrict__ s, _Float16* __restrict__ d, int n) {
    int i = blockIdx.x * blockDim.x + threadIdx.x;
    if (i < n) d[i] = (_Float16)s[i];
}

__global__ void fill_ones_kernel(float* __restrict__ p, int n) {
    int i = blockIdx.x * blockDim.x + threadIdx.x;
    if (i < n) p[i] = 1.0f;
}

__global__ void bn_fold_kernel(const float* __restrict__ g, const float* __restrict__ b,
                               const float* __restrict__ m, const float* __restrict__ v,
                               float* __restrict__ sc, float* __restrict__ bi, int n) {
    int i = blockIdx.x * blockDim.x + threadIdx.x;
    if (i < n) {
        float s = g[i] * rsqrtf(v[i] + 1e-5f);
        sc[i] = s;
        bi[i] = b[i] - m[i] * s;
    }
}

// ---------------------------------------------------------------------------
// depthwise RepVGGDW ( bn(dw3x3) + bn(dw1x1) + identity ) with optional GELU
// ---------------------------------------------------------------------------
__global__ void repvgg_dw_kernel(const float* __restrict__ inA, const float* __restrict__ inB,
                                 const float* __restrict__ inC,
                                 const float* __restrict__ w3,                    // [C,9]
                                 const float* __restrict__ s3, const float* __restrict__ b3,
                                 const float* __restrict__ w1,                    // [C]
                                 const float* __restrict__ s1, const float* __restrict__ b1,
                                 float* __restrict__ out32, _Float16* __restrict__ out16,
                                 int C, int do_gelu) {
    long total = (long)NBATCH * C * PIX;
    for (long idx = blockIdx.x * (long)blockDim.x + threadIdx.x; idx < total;
         idx += (long)gridDim.x * blockDim.x) {
        int  p  = (int)(idx & (PIX - 1));
        long bc = idx >> 12;
        int  c  = (int)(bc % C);
        int  b  = (int)(bc / C);
        int  y  = p >> 6, x = p & 63;
        const float* src; int cc;
        if (c < 256)      { src = inA; cc = c; }
        else if (c < 512) { src = inB; cc = c - 256; }
        else              { src = inC; cc = c - 512; }
        const float* plane = src + ((size_t)b * 256 + cc) * PIX;
        float acc = 0.0f;
#pragma unroll
        for (int dy = -1; dy <= 1; ++dy)
#pragma unroll
            for (int dx = -1; dx <= 1; ++dx) {
                int yy = y + dy, xx = x + dx;
                if (yy >= 0 && yy < 64 && xx >= 0 && xx < 64)
                    acc += w3[c * 9 + (dy + 1) * 3 + (dx + 1)] * plane[yy * 64 + xx];
            }
        float x0 = plane[p];
        float v  = acc * s3[c] + b3[c] + (x0 * w1[c]) * s1[c] + b1[c] + x0;
        if (do_gelu) v = gelu_exact(v);
        if (out32) out32[idx] = v;
        if (out16) out16[idx] = (_Float16)v;
    }
}

// ---------------------------------------------------------------------------
// WMMA GEMM for NCHW 1x1 conv:
//   out[b,m,p] = act( W[m,:] . X[b,:,p] * scale[m] + bias[m] ) (+ resid[b,m,p])
// block = 128 threads (4 waves); tile = 64(M) x 128(pixels); K stepped by 32.
// Each wave: 8 WMMAs per K-step (one A frag reused across 8 N-tiles).
// A tile row-major [64][40]; B tile TRANSPOSED [pixel][k] = [128][40].
// Epilogue fully specialized at compile time; scale/bias hoisted per row.
// ---------------------------------------------------------------------------
template <bool HR, bool O32, bool O16, bool GELU>
__global__ __launch_bounds__(128)
void gemm_nchw_wmma(const _Float16* __restrict__ W,    // [M,K]
                    const _Float16* __restrict__ X,    // [NBATCH,K,PIX]
                    const float* __restrict__ scale,   // [M]
                    const float* __restrict__ bias,    // [M]
                    const float* __restrict__ resid,   // [NBATCH,M,PIX] if HR
                    float* __restrict__ out32,         // if O32
                    _Float16* __restrict__ out16,      // if O16
                    int M, int K) {
    __shared__ __align__(16) _Float16 As [64][40];
    __shared__ __align__(16) _Float16 BsT[128][40];
    const int b  = blockIdx.z;
    const int m0 = blockIdx.y * 64;
    const int n0 = blockIdx.x * 128;
    const int t  = threadIdx.x;
    const int wv = t >> 5;
    const int ln = t & 31;

    v8f acc[8] = {};

    for (int k0 = 0; k0 < K; k0 += 32) {
        {   // A tile 64x32 : thread t loads 16 contiguous halves of one row
            int r = t >> 1, c = (t & 1) << 4;
            const _Float16* g = W + (size_t)(m0 + r) * K + (k0 + c);
            v8h a0 = *(const v8h*)(g);
            v8h a1 = *(const v8h*)(g + 8);
            *(v8h*)&As[r][c]     = a0;
            *(v8h*)&As[r][c + 8] = a1;
        }
#pragma unroll
        for (int e = 0; e < 2; ++e) {
            // B tile 32(K)x128(pix): 16-contiguous-pixel chunks, scattered
            // transposed into BsT[pixel][k]
            int cu = e * 128 + t;
            int k = cu >> 3, n = (cu & 7) << 4;
            const _Float16* g = X + ((size_t)b * K + (k0 + k)) * PIX + (n0 + n);
            v8h b0 = *(const v8h*)(g);
            v8h b1 = *(const v8h*)(g + 8);
#pragma unroll
            for (int i = 0; i < 8; ++i) BsT[n + i][k]     = b0[i];
#pragma unroll
            for (int i = 0; i < 8; ++i) BsT[n + 8 + i][k] = b1[i];
        }
        __syncthreads();
        v16h af = ld_a_frag(&As[wv * 16][0], 40);
#pragma unroll
        for (int j = 0; j < 8; ++j) {
            v16h bf = ld_b_frag_t(&BsT[j * 16][0], 40);
            acc[j] = __builtin_amdgcn_wmma_f32_16x16x32_f16(false, af, false, bf,
                                                            (short)0, acc[j], false, false);
        }
        __syncthreads();
    }

    const int half = ln >> 4;
    const int nc   = ln & 15;
#pragma unroll
    for (int r = 0; r < 8; ++r) {
        int   mg = m0 + wv * 16 + r + half * 8;
        float s  = scale[mg];
        float bb = bias[mg];
        size_t rowo = ((size_t)b * M + mg) * PIX;
#pragma unroll
        for (int j = 0; j < 8; ++j) {
            int ng = n0 + j * 16 + nc;
            float v = acc[j][r] * s + bb;
            if (GELU) v = gelu_exact(v);
            size_t o = rowo + ng;
            if (HR)  v += resid[o];
            if (O32) out32[o] = v;
            if (O16) out16[o] = (_Float16)v;
        }
    }
}

// ---------------------------------------------------------------------------
// Window attention (8x8 windows, 64 tokens, head_dim 64) with WMMA.
// One block (128 thr, 4 waves) per (window, head). Each wave owns a 16-token strip.
// Qs [token][dim]  : A of S=QK^T
// Ks [token][dim]  : transposed-B of S (B = K^T [dim][token])
// Vs [dim][token]  : transposed-B of O=PV (B = V [token][dim])
// Ps [token][token]: A of O
// ---------------------------------------------------------------------------
__global__ __launch_bounds__(128)
void win_attn_wmma(const _Float16* __restrict__ qb, int qC, int qoff,
                   const _Float16* __restrict__ kb, int kC, int koff,
                   const _Float16* __restrict__ vb, int vC, int voff,
                   _Float16* __restrict__ ob /* [NBATCH,256,PIX] */) {
    __shared__ __align__(16) _Float16 Qs[64][72];
    __shared__ __align__(16) _Float16 Ks[64][72];
    __shared__ __align__(16) _Float16 Vs[64][72];
    __shared__ __align__(16) _Float16 Ps[64][72];

    const int wi = blockIdx.x;            // 0..511
    const int h  = blockIdx.y;            // head
    const int b  = wi >> 6;
    const int wy = (wi >> 3) & 7;
    const int wx = wi & 7;
    const int t  = threadIdx.x;
    const int wv = t >> 5;
    const int ln = t & 31;

    // cooperative load: vectorized over the 8 contiguous pixels of a window row
#pragma unroll
    for (int e = 0; e < 4; ++e) {
        int u  = e * 128 + t;             // 512 chunks of 8 halves
        int d  = u >> 3;                  // dim 0..63
        int iy = u & 7;                   // window row
        int n0 = iy << 3;                 // first token of that row
        int p0 = ((wy << 3) + iy) * 64 + (wx << 3);
        v8h qv = *(const v8h*)(qb + ((size_t)b * qC + qoff + h * 64 + d) * PIX + p0);
        v8h kv = *(const v8h*)(kb + ((size_t)b * kC + koff + h * 64 + d) * PIX + p0);
        v8h vv = *(const v8h*)(vb + ((size_t)b * vC + voff + h * 64 + d) * PIX + p0);
#pragma unroll
        for (int i = 0; i < 8; ++i) Qs[n0 + i][d] = qv[i];
#pragma unroll
        for (int i = 0; i < 8; ++i) Ks[n0 + i][d] = kv[i];
        *(v8h*)&Vs[d][n0] = vv;
    }
    __syncthreads();

    // S = Q K^T (strip of 16 rows per wave)
    v8f sacc[4] = {};
#pragma unroll
    for (int kk = 0; kk < 64; kk += 32) {
        v16h af = ld_a_frag(&Qs[wv * 16][kk], 72);
#pragma unroll
        for (int j = 0; j < 4; ++j) {
            v16h bf = ld_b_frag_t(&Ks[j * 16][kk], 72);
            sacc[j] = __builtin_amdgcn_wmma_f32_16x16x32_f16(false, af, false, bf,
                                                             (short)0, sacc[j], false, false);
        }
    }

    const int half = ln >> 4;
    const int nc   = ln & 15;
    // softmax: row elements live in one 16-lane group (lanes 0-15 = row r,
    // lanes 16-31 = row r+8) across the 4 j-tiles.
#pragma unroll
    for (int r = 0; r < 8; ++r) {
        float mx = -3.4e38f;
#pragma unroll
        for (int j = 0; j < 4; ++j) {
            sacc[j][r] = sacc[j][r] * 0.125f;   // hd^-0.5, hd=64
            mx = fmaxf(mx, sacc[j][r]);
        }
#pragma unroll
        for (int mk = 8; mk >= 1; mk >>= 1) mx = fmaxf(mx, __shfl_xor(mx, mk, 32));
        float ev[4];
        float sum = 0.0f;
#pragma unroll
        for (int j = 0; j < 4; ++j) { ev[j] = __expf(sacc[j][r] - mx); sum += ev[j]; }
#pragma unroll
        for (int mk = 8; mk >= 1; mk >>= 1) sum += __shfl_xor(sum, mk, 32);
        float inv = 1.0f / sum;
        int row = wv * 16 + r + half * 8;
#pragma unroll
        for (int j = 0; j < 4; ++j) Ps[row][j * 16 + nc] = (_Float16)(ev[j] * inv);
    }
    __syncthreads();

    // O = P V
    v8f oacc[4] = {};
#pragma unroll
    for (int kk = 0; kk < 64; kk += 32) {
        v16h af = ld_a_frag(&Ps[wv * 16][kk], 72);
#pragma unroll
        for (int j = 0; j < 4; ++j) {
            v16h bf = ld_b_frag_t(&Vs[j * 16][kk], 72);
            oacc[j] = __builtin_amdgcn_wmma_f32_16x16x32_f16(false, af, false, bf,
                                                             (short)0, oacc[j], false, false);
        }
    }

#pragma unroll
    for (int j = 0; j < 4; ++j) {
#pragma unroll
        for (int r = 0; r < 8; ++r) {
            int n = wv * 16 + r + half * 8;
            int d = j * 16 + nc;
            int p = ((wy << 3) + (n >> 3)) * 64 + (wx << 3) + (n & 7);
            ob[((size_t)b * 256 + h * 64 + d) * PIX + p] = (_Float16)oacc[j][r];
        }
    }
}

// ---------------------------------------------------------------------------
// SE: channel mean, tiny fc, hardsigmoid gate
// ---------------------------------------------------------------------------
__global__ void chan_mean_kernel(const float* __restrict__ in, float* __restrict__ mean) {
    __shared__ float red[256];
    int bc = blockIdx.x;
    const float* p = in + (size_t)bc * PIX;
    float s = 0.0f;
    for (int i = threadIdx.x; i < PIX; i += 256) s += p[i];
    red[threadIdx.x] = s;
    __syncthreads();
    for (int st = 128; st > 0; st >>= 1) {
        if (threadIdx.x < st) red[threadIdx.x] += red[threadIdx.x + st];
        __syncthreads();
    }
    if (threadIdx.x == 0) mean[bc] = red[0] * (1.0f / PIX);
}

__global__ void se_fc_kernel(const float* __restrict__ mean, const float* __restrict__ w,
                             const float* __restrict__ bias, float* __restrict__ se) {
    int b = blockIdx.x, co = threadIdx.x;
    float s = bias[co];
    for (int ci = 0; ci < 256; ++ci) s += w[co * 256 + ci] * mean[b * 256 + ci];
    se[b * 256 + co] = s;
}

__global__ void se_gate_kernel(const float* __restrict__ llx, const float* __restrict__ se,
                               float* __restrict__ o32, _Float16* __restrict__ o16) {
    long total = (long)NBATCH * 256 * PIX;
    for (long idx = blockIdx.x * (long)blockDim.x + threadIdx.x; idx < total;
         idx += (long)gridDim.x * blockDim.x) {
        long  bc = idx >> 12;
        float g  = se[bc] * (1.0f / 6.0f) + 0.5f;
        g = fminf(fmaxf(g, 0.0f), 1.0f);
        float v = llx[idx] * g;
        o32[idx] = v;
        o16[idx] = (_Float16)v;
    }
}

// ---------------------------------------------------------------------------
// host launch
// ---------------------------------------------------------------------------
extern "C" void kernel_launch(void* const* d_in, const int* in_sizes, int n_in,
                              void* d_out, int out_size, void* d_ws, size_t ws_size,
                              hipStream_t stream) {
    (void)in_sizes; (void)n_in; (void)out_size; (void)ws_size;
    const float* ll = (const float*)d_in[0];
    const float* lh = (const float*)d_in[1];
    const float* hl = (const float*)d_in[2];
    const float* hh = (const float*)d_in[3];
    int ii = 4;
    auto nx = [&]() { return (const float*)d_in[ii++]; };
    const float *hf_dw3_w = nx(), *hf_dw3_g = nx(), *hf_dw3_b = nx(), *hf_dw3_m = nx(), *hf_dw3_v = nx();
    const float *hf_dw1_w = nx(), *hf_dw1_g = nx(), *hf_dw1_b = nx(), *hf_dw1_m = nx(), *hf_dw1_v = nx();
    const float *hf_pw_w  = nx(), *hf_pw_g  = nx(), *hf_pw_b  = nx(), *hf_pw_m  = nx(), *hf_pw_v  = nx();
    const float *ha_qkv_w = nx(), *ha_qkv_b = nx();
    const float *ha_proj_w = nx(), *ha_proj_b = nx();
    const float *hm_pw1_w = nx(), *hm_pw1_g = nx(), *hm_pw1_b = nx(), *hm_pw1_m = nx(), *hm_pw1_v = nx();
    const float *hm_pw2_w = nx(), *hm_pw2_g = nx(), *hm_pw2_b = nx(), *hm_pw2_m = nx(), *hm_pw2_v = nx();
    const float *ld_dw3_w = nx(), *ld_dw3_g = nx(), *ld_dw3_b = nx(), *ld_dw3_m = nx(), *ld_dw3_v = nx();
    const float *ld_dw1_w = nx(), *ld_dw1_g = nx(), *ld_dw1_b = nx(), *ld_dw1_m = nx(), *ld_dw1_v = nx();
    const float *le_fc_w = nx(), *le_fc_b = nx();
    const float *la_q_w = nx(), *la_q_b = nx();
    const float *la_kv_w = nx(), *la_kv_b = nx();
    const float *la_p_w = nx(), *la_p_b = nx();
    const float *lm_pw1_w = nx(), *lm_pw1_g = nx(), *lm_pw1_b = nx(), *lm_pw1_m = nx(), *lm_pw1_v = nx();
    const float *lm_pw2_w = nx(), *lm_pw2_g = nx(), *lm_pw2_b = nx(), *lm_pw2_m = nx(), *lm_pw2_v = nx();

    char* base = (char*)d_ws;
    size_t off = 0;
    auto alloc = [&](size_t bytes) -> void* {
        void* p = base + off;
        off = (off + bytes + 255) & ~(size_t)255;
        return p;
    };

    // f16 weights
    _Float16* w_pw16  = (_Float16*)alloc((size_t)256 * 768 * 2);
    _Float16* w_qkv16 = (_Float16*)alloc((size_t)768 * 256 * 2);
    _Float16* w_prj16 = (_Float16*)alloc((size_t)256 * 256 * 2);
    _Float16* w_hm116 = (_Float16*)alloc((size_t)512 * 256 * 2);
    _Float16* w_hm216 = (_Float16*)alloc((size_t)256 * 512 * 2);
    _Float16* w_laq16 = (_Float16*)alloc((size_t)256 * 256 * 2);
    _Float16* w_lakv16= (_Float16*)alloc((size_t)512 * 256 * 2);
    _Float16* w_lap16 = (_Float16*)alloc((size_t)256 * 256 * 2);
    _Float16* w_lm116 = (_Float16*)alloc((size_t)512 * 256 * 2);
    _Float16* w_lm216 = (_Float16*)alloc((size_t)256 * 512 * 2);
    // folded BN scale/bias + ones vector (unit scale for bias-only convs)
    float* ones  = (float*)alloc(768 * 4);
    float* hf3_s = (float*)alloc(768 * 4); float* hf3_b = (float*)alloc(768 * 4);
    float* hf1_s = (float*)alloc(768 * 4); float* hf1_b = (float*)alloc(768 * 4);
    float* pw_s  = (float*)alloc(256 * 4); float* pw_b  = (float*)alloc(256 * 4);
    float* hm1_s = (float*)alloc(512 * 4); float* hm1_b = (float*)alloc(512 * 4);
    float* hm2_s = (float*)alloc(256 * 4); float* hm2_b = (float*)alloc(256 * 4);
    float* ld3_s = (float*)alloc(256 * 4); float* ld3_b = (float*)alloc(256 * 4);
    float* ld1_s = (float*)alloc(256 * 4); float* ld1_b = (float*)alloc(256 * 4);
    float* lm1_s = (float*)alloc(512 * 4); float* lm1_b = (float*)alloc(512 * 4);
    float* lm2_s = (float*)alloc(256 * 4); float* lm2_b = (float*)alloc(256 * 4);
    // activations
    size_t act256_16 = (size_t)NBATCH * 256 * PIX * 2;
    size_t act256_32 = (size_t)NBATCH * 256 * PIX * 4;
    size_t act512_16 = (size_t)NBATCH * 512 * PIX * 2;
    size_t act768_16 = (size_t)NBATCH * 768 * PIX * 2;
    _Float16* high16  = (_Float16*)alloc(act768_16);   // gelu(repvgg(concat))
    float*    hc32    = (float*)alloc(act256_32);
    _Float16* hc16    = (_Float16*)alloc(act256_16);
    _Float16* qkv16   = (_Float16*)alloc(act768_16);
    _Float16* attn16  = (_Float16*)alloc(act256_16);
    float*    hat32   = (float*)alloc(act256_32);      // high_attn_out
    _Float16* hat16   = (_Float16*)alloc(act256_16);
    _Float16* ffh16   = (_Float16*)alloc(act512_16);
    float*    ref32   = (float*)alloc(act256_32);      // refined_high
    _Float16* ref16   = (_Float16*)alloc(act256_16);
    float*    llx32   = (float*)alloc(act256_32);
    float*    semean  = (float*)alloc(NBATCH * 256 * 4);
    float*    seout   = (float*)alloc(NBATCH * 256 * 4);
    float*    llese32 = (float*)alloc(act256_32);
    _Float16* llese16 = (_Float16*)alloc(act256_16);
    _Float16* q16     = (_Float16*)alloc(act256_16);
    _Float16* kv16    = (_Float16*)alloc(act512_16);
    _Float16* cat16   = (_Float16*)alloc(act256_16);
    float*    lat32   = (float*)alloc(act256_32);      // ll_attn
    _Float16* lat16   = (_Float16*)alloc(act256_16);
    _Float16* lmh16   = (_Float16*)alloc(act512_16);

    auto cvt = [&](const float* s, _Float16* d, int n) {
        cvt_f16_kernel<<<dim3((n + 255) / 256), dim3(256), 0, stream>>>(s, d, n);
    };
    auto fold = [&](const float* g, const float* b, const float* m, const float* v,
                    float* sc, float* bi, int n) {
        bn_fold_kernel<<<dim3((n + 255) / 256), dim3(256), 0, stream>>>(g, b, m, v, sc, bi, n);
    };
    const dim3 gthr(128);
    auto ggrid = [&](int M) { return dim3(PIX / 128, M / 64, NBATCH); };

    // -------- prep: f16 weights + BN folds --------
    fill_ones_kernel<<<dim3(3), dim3(256), 0, stream>>>(ones, 768);
    cvt(hf_pw_w,  w_pw16,  256 * 768);
    cvt(ha_qkv_w, w_qkv16, 768 * 256);
    cvt(ha_proj_w,w_prj16, 256 * 256);
    cvt(hm_pw1_w, w_hm116, 512 * 256);
    cvt(hm_pw2_w, w_hm216, 256 * 512);
    cvt(la_q_w,   w_laq16, 256 * 256);
    cvt(la_kv_w,  w_lakv16,512 * 256);
    cvt(la_p_w,   w_lap16, 256 * 256);
    cvt(lm_pw1_w, w_lm116, 512 * 256);
    cvt(lm_pw2_w, w_lm216, 256 * 512);
    fold(hf_dw3_g, hf_dw3_b, hf_dw3_m, hf_dw3_v, hf3_s, hf3_b, 768);
    fold(hf_dw1_g, hf_dw1_b, hf_dw1_m, hf_dw1_v, hf1_s, hf1_b, 768);
    fold(hf_pw_g,  hf_pw_b,  hf_pw_m,  hf_pw_v,  pw_s,  pw_b,  256);
    fold(hm_pw1_g, hm_pw1_b, hm_pw1_m, hm_pw1_v, hm1_s, hm1_b, 512);
    fold(hm_pw2_g, hm_pw2_b, hm_pw2_m, hm_pw2_v, hm2_s, hm2_b, 256);
    fold(ld_dw3_g, ld_dw3_b, ld_dw3_m, ld_dw3_v, ld3_s, ld3_b, 256);
    fold(ld_dw1_g, ld_dw1_b, ld_dw1_m, ld_dw1_v, ld1_s, ld1_b, 256);
    fold(lm_pw1_g, lm_pw1_b, lm_pw1_m, lm_pw1_v, lm1_s, lm1_b, 512);
    fold(lm_pw2_g, lm_pw2_b, lm_pw2_m, lm_pw2_v, lm2_s, lm2_b, 256);

    // -------- high-frequency branch --------
    repvgg_dw_kernel<<<dim3(4096), dim3(256), 0, stream>>>(
        lh, hl, hh, hf_dw3_w, hf3_s, hf3_b, hf_dw1_w, hf1_s, hf1_b,
        nullptr, high16, 768, 1);
    // high_compressed = BN(pw(high))
    gemm_nchw_wmma<false, true, true, false><<<ggrid(256), gthr, 0, stream>>>(
        w_pw16, high16, pw_s, pw_b, nullptr, hc32, hc16, 256, 768);
    // qkv = conv(hc) + b
    gemm_nchw_wmma<false, false, true, false><<<ggrid(768), gthr, 0, stream>>>(
        w_qkv16, hc16, ones, ha_qkv_b, nullptr, nullptr, qkv16, 768, 256);
    win_attn_wmma<<<dim3(512, 4), dim3(128), 0, stream>>>(
        qkv16, 768, 0, qkv16, 768, 256, qkv16, 768, 512, attn16);
    // proj + residual hc
    gemm_nchw_wmma<true, true, true, false><<<ggrid(256), gthr, 0, stream>>>(
        w_prj16, attn16, ones, ha_proj_b, hc32, hat32, hat16, 256, 256);
    // FFN
    gemm_nchw_wmma<false, false, true, true><<<ggrid(512), gthr, 0, stream>>>(
        w_hm116, hat16, hm1_s, hm1_b, nullptr, nullptr, ffh16, 512, 256);
    gemm_nchw_wmma<true, true, true, false><<<ggrid(256), gthr, 0, stream>>>(
        w_hm216, ffh16, hm2_s, hm2_b, hat32, ref32, ref16, 256, 512);

    // -------- low-frequency branch --------
    repvgg_dw_kernel<<<dim3(2048), dim3(256), 0, stream>>>(
        ll, ll, ll, ld_dw3_w, ld3_s, ld3_b, ld_dw1_w, ld1_s, ld1_b,
        llx32, nullptr, 256, 0);
    chan_mean_kernel<<<dim3(NBATCH * 256), dim3(256), 0, stream>>>(llx32, semean);
    se_fc_kernel<<<dim3(NBATCH), dim3(256), 0, stream>>>(semean, le_fc_w, le_fc_b, seout);
    se_gate_kernel<<<dim3(2048), dim3(256), 0, stream>>>(llx32, seout, llese32, llese16);

    gemm_nchw_wmma<false, false, true, false><<<ggrid(256), gthr, 0, stream>>>(
        w_laq16, llese16, ones, la_q_b, nullptr, nullptr, q16, 256, 256);
    gemm_nchw_wmma<false, false, true, false><<<ggrid(512), gthr, 0, stream>>>(
        w_lakv16, ref16, ones, la_kv_b, nullptr, nullptr, kv16, 512, 256);
    win_attn_wmma<<<dim3(512, 4), dim3(128), 0, stream>>>(
        q16, 256, 0, kv16, 512, 0, kv16, 512, 256, cat16);
    gemm_nchw_wmma<true, true, true, false><<<ggrid(256), gthr, 0, stream>>>(
        w_lap16, cat16, ones, la_p_b, llese32, lat32, lat16, 256, 256);
    gemm_nchw_wmma<false, false, true, true><<<ggrid(512), gthr, 0, stream>>>(
        w_lm116, lat16, lm1_s, lm1_b, nullptr, nullptr, lmh16, 512, 256);
    gemm_nchw_wmma<true, true, false, false><<<ggrid(256), gthr, 0, stream>>>(
        w_lm216, lmh16, lm2_s, lm2_b, lat32, (float*)d_out, nullptr, 256, 512);
}
